// MixLinear_FP8GEMM_33071248179802
// MI455X (gfx1250) — 8-bit matrix operand order repaired and checked by reading, compiled, not run on hardware
//
#include <hip/hip_runtime.h>
#include <hip/hip_fp16.h>

typedef __attribute__((ext_vector_type(16))) int   v16i;
typedef __attribute__((ext_vector_type(8)))  float v8f;

#define FP8_MAX 448.0f
#define TILE 128

#if __has_builtin(__builtin_amdgcn_global_load_async_to_lds_b128) && \
    __has_builtin(__builtin_amdgcn_s_wait_asynccnt)
#define USE_ASYNC_LDS 1
typedef int b128_t __attribute__((vector_size(4 * sizeof(int))));
typedef __attribute__((address_space(1))) b128_t* gptr128_t;
typedef __attribute__((address_space(3))) b128_t* lptr128_t;
#else
#define USE_ASYNC_LDS 0
#endif


__device__ inline unsigned cvt_e4m3_sw(float x) {
    unsigned b = __float_as_uint(x);
    unsigned sign = (b >> 24) & 0x80u;
    b &= 0x7FFFFFFFu;
    if (b >= 0x43E00000u) return sign | 0x7Eu;
    if (b < 0x3C800000u) {
        unsigned m = (unsigned)__float2int_rn(__uint_as_float(b) * 512.0f);
        return sign | m;
    }
    unsigned lsb = (b >> 20) & 1u;
    b += 0x7FFFFu + lsb;
    unsigned e = (b >> 23) - 120u;
    unsigned m = (b >> 20) & 7u;
    if (e > 15u || (e == 15u && m > 6u)) return sign | 0x7Eu;
    return sign | (e << 3) | m;
}

__device__ inline unsigned pack2_fp8(float a, float b) {
#if __has_builtin(__builtin_amdgcn_cvt_pk_fp8_f32)
    return (unsigned)__builtin_amdgcn_cvt_pk_fp8_f32(a, b, 0, false) & 0xFFFFu;
#else
    return (cvt_e4m3_sw(a) & 0xFFu) | ((cvt_e4m3_sw(b) & 0xFFu) << 8);
#endif
}


__global__ __launch_bounds__(256) void absmax_kernel(
    const unsigned* __restrict__ x2, long n, unsigned* __restrict__ out) {
    long i = (long)blockIdx.x * blockDim.x + threadIdx.x;
    long stride = (long)gridDim.x * blockDim.x;
    float m = 0.0f;
    for (; i < n; i += stride) {
        unsigned u = x2[i] & 0x7FFF7FFFu;
        __half2 h = *reinterpret_cast<__half2*>(&u);
        float2 f = __half22float2(h);
        m = fmaxf(m, fmaxf(f.x, f.y));
    }
    #pragma unroll
    for (int off = 16; off > 0; off >>= 1)
        m = fmaxf(m, __shfl_down(m, off, 32));
    __shared__ float sm[8];
    int lane = threadIdx.x & 31, wv = threadIdx.x >> 5;
    if (lane == 0) sm[wv] = m;
    __syncthreads();
    if (wv == 0) {
        m = (lane < 8) ? sm[lane] : 0.0f;
        #pragma unroll
        for (int off = 4; off > 0; off >>= 1)
            m = fmaxf(m, __shfl_down(m, off, 32));
        if (lane == 0) atomicMax(out, __float_as_uint(m));
    }
}


__global__ __launch_bounds__(256) void quant4_kernel(
    const __half2* __restrict__ xp, unsigned* __restrict__ q, long n4,
    const unsigned* __restrict__ maxbits, int scaled) {
    float inv = 1.0f;
    if (scaled) {
        float mx = __uint_as_float(*maxbits);
        inv = mx > 0.0f ? FP8_MAX / mx : 0.0f;
    }
    long i = (long)blockIdx.x * blockDim.x + threadIdx.x;
    long stride = (long)gridDim.x * blockDim.x;
    for (; i < n4; i += stride) {
        float2 f0 = __half22float2(xp[2 * i]);
        float2 f1 = __half22float2(xp[2 * i + 1]);
        float a = fminf(fmaxf(f0.x * inv, -FP8_MAX), FP8_MAX);
        float b = fminf(fmaxf(f0.y * inv, -FP8_MAX), FP8_MAX);
        float c = fminf(fmaxf(f1.x * inv, -FP8_MAX), FP8_MAX);
        float d = fminf(fmaxf(f1.y * inv, -FP8_MAX), FP8_MAX);
        q[i] = pack2_fp8(a, b) | (pack2_fp8(c, d) << 16);
    }
}


__device__ inline void compute_ktile(const unsigned char* __restrict__ A,
                                     const unsigned char* __restrict__ B,
                                     v8f acc[8], int lane, int wave) {
    const int am    = wave * 16 + (lane & 15);
    const int khalf = (lane >> 4) << 3;
    v16i afrag;
    #pragma unroll
    for (int v = 0; v < 8; v++) {
        int kb = ((v >> 1) << 4) + ((v & 1) << 2) + khalf;
        afrag[v]     = *(const int*)&A[am * TILE + kb];
        afrag[v + 8] = *(const int*)&A[am * TILE + kb + 64];
    }
    #pragma unroll
    for (int nt = 0; nt < 8; nt++) {
        const int bn = nt * 16 + (lane & 15);
        v16i bfrag;
        #pragma unroll
        for (int j = 0; j < 16; j++) {
            int kb = (((j & 7) >> 1) << 4) + ((j & 1) << 2) + ((lane >> 4) << 3) + ((j >> 3) << 6);
            bfrag[j] = *(const int*)&B[bn * TILE + kb];
        }
        acc[nt] = __builtin_amdgcn_wmma_f32_16x16x128_fp8_fp8(
            afrag, bfrag, (short)0, acc[nt], false, false);
    }
}

#if USE_ASYNC_LDS
__device__ inline void issue_ktile_async(const unsigned char* __restrict__ qx,
                                         const unsigned char* __restrict__ qw,
                                         unsigned char* bufA, unsigned char* bufB,
                                         int row0, int col0, int K, int k0, int tid) {
    #pragma unroll
    for (int i = 0; i < 4; i++) {
        int idx = tid + i * 256;
        int r = idx >> 3, c = idx & 7;
        __builtin_amdgcn_global_load_async_to_lds_b128(
            (gptr128_t)(qx + (size_t)(row0 + r) * K + k0 + c * 16),
            (lptr128_t)(bufA + idx * 16), 0, 0);
        __builtin_amdgcn_global_load_async_to_lds_b128(
            (gptr128_t)(qw + (size_t)(col0 + r) * K + k0 + c * 16),
            (lptr128_t)(bufB + idx * 16), 0, 0);
    }
}
#endif

__global__ __launch_bounds__(256) void gemm_fp8_kernel(
    const unsigned char* __restrict__ qx, const unsigned char* __restrict__ qw,
    const unsigned* __restrict__ maxbits, const __half* __restrict__ bias,
    __half* __restrict__ out, int M, int N, int K) {
#if USE_ASYNC_LDS
    __shared__ unsigned char ldsA[2][TILE * TILE];
    __shared__ unsigned char ldsB[2][TILE * TILE];
#else
    __shared__ unsigned char ldsA[1][TILE * TILE];
    __shared__ unsigned char ldsB[1][TILE * TILE];
#endif

    const int tid  = threadIdx.x;
    const int lane = tid & 31;
    const int wave = tid >> 5;
    const int row0 = blockIdx.x * TILE;
    const int col0 = blockIdx.y * TILE;

    const float mx    = __uint_as_float(*maxbits);
    const float scale = mx / FP8_MAX;

    v8f acc[8];
    #pragma unroll
    for (int i = 0; i < 8; i++)
        #pragma unroll
        for (int j = 0; j < 8; j++) acc[i][j] = 0.0f;

#if USE_ASYNC_LDS
    const int KT = K / TILE;
    issue_ktile_async(qx, qw, ldsA[0], ldsB[0], row0, col0, K, 0, tid);
    for (int kt = 0; kt < KT; kt++) {
        const int cur = kt & 1;
        if (kt + 1 < KT) {
            issue_ktile_async(qx, qw, ldsA[cur ^ 1], ldsB[cur ^ 1],
                              row0, col0, K, (kt + 1) * TILE, tid);
            __builtin_amdgcn_s_wait_asynccnt(8);
        } else {
            __builtin_amdgcn_s_wait_asynccnt(0);
        }
        __syncthreads();
        compute_ktile(ldsA[cur], ldsB[cur], acc, lane, wave);
        __syncthreads();
    }
#else
    for (int k0 = 0; k0 < K; k0 += TILE) {
        #pragma unroll
        for (int i = 0; i < 4; i++) {
            int idx = tid + i * 256;
            int r = idx >> 3, c = idx & 7;
            ((uint4*)ldsA[0])[idx] = *(const uint4*)(qx + (size_t)(row0 + r) * K + k0 + c * 16);
            ((uint4*)ldsB[0])[idx] = *(const uint4*)(qw + (size_t)(col0 + r) * K + k0 + c * 16);
        }
        __syncthreads();
        compute_ktile(ldsA[0], ldsB[0], acc, lane, wave);
        __syncthreads();
    }
#endif

    #pragma unroll
    for (int nt = 0; nt < 8; nt++) {
        const int col = col0 + nt * 16 + (lane & 15);
        const float bv = __half2float(bias[col]);
        #pragma unroll
        for (int r = 0; r < 8; r++) {
            const int m = wave * 16 + r + ((lane >> 4) << 3);
            out[(size_t)(row0 + m) * N + col] = __float2half(acc[nt][r] * scale + bv);
        }
    }
}


extern "C" void kernel_launch(void* const* d_in, const int* in_sizes, int n_in,
                              void* d_out, int out_size, void* d_ws, size_t ws_size,
                              hipStream_t stream) {
    const __half* x    = (const __half*)d_in[0];
    const __half* w    = (const __half*)d_in[1];
    const __half* bias = (const __half*)d_in[2];
    __half* out        = (__half*)d_out;

    const int N = in_sizes[2];
    const int K = in_sizes[1] / N;
    const int M = in_sizes[0] / K;

    unsigned char* ws = (unsigned char*)d_ws;
    unsigned* scale_bits = (unsigned*)ws;
    unsigned char* qx = ws + 256;
    unsigned char* qw = qx + (size_t)M * K;

    (void)hipMemsetAsync(d_ws, 0, 256, stream);

    absmax_kernel<<<1024, 256, 0, stream>>>(
        (const unsigned*)x, (long)M * K / 2, scale_bits);

    quant4_kernel<<<512, 256, 0, stream>>>(
        (const __half2*)w, (unsigned*)qw, (long)N * K / 4, scale_bits, 0);

    quant4_kernel<<<2048, 256, 0, stream>>>(
        (const __half2*)x, (unsigned*)qx, (long)M * K / 4, scale_bits, 1);

    dim3 grid(M / TILE, N / TILE);
    gemm_fp8_kernel<<<grid, 256, 0, stream>>>(qx, qw, scale_bits, bias, out, M, N, K);
}
